// MultiHeadSelfAttention_88587995447761
// MI455X (gfx1250) — compile-verified, matrix operand order checked by reading, not run on hardware
//
#include <hip/hip_runtime.h>
#include <hip/hip_bf16.h>
#include <stdint.h>

typedef __attribute__((ext_vector_type(16))) __bf16 v16bf;
typedef __attribute__((ext_vector_type(8)))  float  v8f;

#define S_LEN  2048
#define DMODEL 1024
#define NHEADS 16
#define HDIM   64
#define MROWS  4096  // B*S

// Load a 16x32 bf16 tile from row-major memory in the WMMA A-operand layout
// (also valid for the B operand when the tile is the [N][K] view of B).
// lanes 0-15: row = lane, K = {0..7, 16..23}; lanes 16-31: same row, K = {8..15, 24..31}.
__device__ __forceinline__ v16bf loadA16(const __bf16* base, int ldRow, int lane) {
  const __bf16* p = base + (size_t)(lane & 15) * ldRow + ((lane & 16) ? 8 : 0);
  union { uint4 u[2]; v16bf v; } r;
  r.u[0] = *reinterpret_cast<const uint4*>(p);
  r.u[1] = *reinterpret_cast<const uint4*>(p + 16);
  return r.v;
}

__device__ __forceinline__ v8f wmma_bf16(v16bf a, v16bf b, v8f c) {
  return __builtin_amdgcn_wmma_f32_16x16x32_bf16(false, a, false, b, (short)0, c,
                                                 false, false);
}

// ---- DPP16 row (16-lane) butterflies: pure VALU, no LDS, no dscnt waits. ----
template <int CTRL>
__device__ __forceinline__ float dpp_mov_f32(float v) {
  int x = __builtin_bit_cast(int, v);
  int y = __builtin_amdgcn_update_dpp(x, x, CTRL, 0xf, 0xf, false);
  return __builtin_bit_cast(float, y);
}
// full reduction within each 16-lane row; all lanes receive the result.
__device__ __forceinline__ void rowreduce_max8(float t[8]) {
#pragma unroll
  for (int g = 0; g < 8; ++g) t[g] = fmaxf(t[g], dpp_mov_f32<0x0B1>(t[g]));
#pragma unroll
  for (int g = 0; g < 8; ++g) t[g] = fmaxf(t[g], dpp_mov_f32<0x04E>(t[g]));
#pragma unroll
  for (int g = 0; g < 8; ++g) t[g] = fmaxf(t[g], dpp_mov_f32<0x141>(t[g]));
#pragma unroll
  for (int g = 0; g < 8; ++g) t[g] = fmaxf(t[g], dpp_mov_f32<0x140>(t[g]));
}
__device__ __forceinline__ void rowreduce_sum8(float t[8]) {
#pragma unroll
  for (int g = 0; g < 8; ++g) t[g] += dpp_mov_f32<0x0B1>(t[g]);
#pragma unroll
  for (int g = 0; g < 8; ++g) t[g] += dpp_mov_f32<0x04E>(t[g]);
#pragma unroll
  for (int g = 0; g < 8; ++g) t[g] += dpp_mov_f32<0x141>(t[g]);
#pragma unroll
  for (int g = 0; g < 8; ++g) t[g] += dpp_mov_f32<0x140>(t[g]);
}

__global__ void cast_f32_bf16(const float* __restrict__ src,
                              __bf16* __restrict__ dst, int n) {
  for (int i = blockIdx.x * blockDim.x + threadIdx.x; i < n;
       i += gridDim.x * blockDim.x)
    dst[i] = (__bf16)src[i];
}

// C[4096,1024] = A[4096,1024](bf16) @ W[1024,1024]^T (W row-major [N][K]) + bias
// Each wave owns a 16x64 tile. MODE 0: bf16 [B,H,S,Hd]; MODE 1: bf16 [B,H,Hd,S]
// (transposed, for V); MODE 2: f32 row-major (final output).
template <int MODE>
__global__ void gemm_bias(const __bf16* __restrict__ A, const __bf16* __restrict__ W,
                          const float* __restrict__ bias, void* __restrict__ out) {
  const int lane = threadIdx.x & 31;
  const int wave = threadIdx.x >> 5;
  const int id = blockIdx.x * 4 + wave;  // 0..4095
  const int mtile = id >> 4;             // 0..255
  const int ngrp  = id & 15;             // 0..15
  const __bf16* Ab = A + (size_t)mtile * 16 * DMODEL;
  const __bf16* Wb = W + (size_t)ngrp * 64 * DMODEL;

  v8f c[4];
#pragma unroll
  for (int t = 0; t < 4; ++t) c[t] = (v8f){0, 0, 0, 0, 0, 0, 0, 0};

#pragma unroll 2
  for (int kk = 0; kk < DMODEL; kk += 32) {
    if (kk + 64 < DMODEL) {  // prefetch next 128B line of each row we stream
      __builtin_prefetch(Ab + (size_t)(lane & 15) * DMODEL + kk + 64, 0, 3);
#pragma unroll
      for (int t = 0; t < 4; ++t)
        __builtin_prefetch(Wb + (size_t)(t * 16 + (lane & 15)) * DMODEL + kk + 64,
                           0, 3);
    }
    v16bf a = loadA16(Ab + kk, DMODEL, lane);
#pragma unroll
    for (int t = 0; t < 4; ++t) {
      v16bf bt = loadA16(Wb + (size_t)t * 16 * DMODEL + kk, DMODEL, lane);
      c[t] = wmma_bf16(a, bt, c[t]);
    }
  }

#pragma unroll
  for (int t = 0; t < 4; ++t) {
    const int n = ngrp * 64 + t * 16 + (lane & 15);
    const float bv = bias[n];
    const int h = n >> 6, hd = n & 63;
#pragma unroll
    for (int g = 0; g < 8; ++g) {
      const int mrow = g + ((lane >> 4) << 3);
      const int mg = mtile * 16 + mrow;
      const int bb = mg >> 11, ss = mg & (S_LEN - 1);
      const float val = c[t][g] + bv;
      if (MODE == 0) {
        ((__bf16*)out)[(((size_t)bb * NHEADS + h) * S_LEN + ss) * HDIM + hd] =
            (__bf16)val;
      } else if (MODE == 1) {
        ((__bf16*)out)[(((size_t)bb * NHEADS + h) * HDIM + hd) * S_LEN + ss] =
            (__bf16)val;
      } else {
        ((float*)out)[(size_t)mg * DMODEL + n] = val;
      }
    }
  }
}

// One wave per 16-query tile; 64 keys per iteration (online softmax).
// Q,K: bf16 [B,H,S,64]; VT: bf16 [B,H,64,S]; CTX out: bf16 [B,S,1024].
__global__ void flash_attn(const __bf16* __restrict__ Q, const __bf16* __restrict__ K,
                           const __bf16* __restrict__ VT, __bf16* __restrict__ CTX) {
  __shared__ alignas(16) __bf16 ptile[4][16 * 64];  // per-wave P tile (2KB each)
  const int lane = threadIdx.x & 31;
  const int wave = threadIdx.x >> 5;
  const int bh = blockIdx.y;                 // 0..31  (b*16 + h)
  const int qtile = blockIdx.x * 4 + wave;   // 0..127
  const int q0 = qtile * 16;

  const __bf16* Qb = Q + ((size_t)bh * S_LEN + q0) * HDIM;
  v16bf qa0 = loadA16(Qb, HDIM, lane);        // hd 0..31
  v16bf qa1 = loadA16(Qb + 32, HDIM, lane);   // hd 32..63

  v8f acc[4];
#pragma unroll
  for (int t = 0; t < 4; ++t) acc[t] = (v8f){0, 0, 0, 0, 0, 0, 0, 0};
  float mrun[8], lrun[8];
#pragma unroll
  for (int g = 0; g < 8; ++g) { mrun[g] = -3.0e38f; lrun[g] = 0.0f; }

  const float K2 = 0.18033688011112042f;  // log2(e) / sqrt(64)
  const v8f zc = (v8f){0, 0, 0, 0, 0, 0, 0, 0};

  for (int j = 0; j < S_LEN; j += 64) {
    if (j + 64 < S_LEN) {  // prefetch next K rows + next VT row segments
      const __bf16* Kn = K + ((size_t)bh * S_LEN + j + 64) * HDIM;
      __builtin_prefetch(Kn + (size_t)lane * HDIM, 0, 3);
      __builtin_prefetch(Kn + (size_t)(lane + 32) * HDIM, 0, 3);
      const __bf16* Vn = VT + (size_t)bh * HDIM * S_LEN + j + 64;
      __builtin_prefetch(Vn + (size_t)(2 * lane) * S_LEN, 0, 3);
      __builtin_prefetch(Vn + (size_t)(2 * lane + 1) * S_LEN, 0, 3);
    }

    const __bf16* Kb = K + ((size_t)bh * S_LEN + j) * HDIM;
    v8f sc[4];
#pragma unroll
    for (int s = 0; s < 4; ++s) {  // 4 key sub-tiles of 16
      v16bf k0 = loadA16(Kb + (size_t)s * 16 * HDIM, HDIM, lane);
      v16bf k1 = loadA16(Kb + (size_t)s * 16 * HDIM + 32, HDIM, lane);
      sc[s] = wmma_bf16(qa1, k1, wmma_bf16(qa0, k0, zc));
    }

    float tmax[8];
#pragma unroll
    for (int g = 0; g < 8; ++g)
      tmax[g] = fmaxf(fmaxf(sc[0][g], sc[1][g]), fmaxf(sc[2][g], sc[3][g]));
    rowreduce_max8(tmax);

    float scl[8], tsum[8];
#pragma unroll
    for (int g = 0; g < 8; ++g) {
      const float mnew = fmaxf(mrun[g], tmax[g]);
      scl[g] = __builtin_amdgcn_exp2f((mrun[g] - mnew) * K2);
      mrun[g] = mnew;
      const int row = g + ((lane >> 4) << 3);
      float psum = 0.0f;
#pragma unroll
      for (int s = 0; s < 4; ++s) {
        const float p = __builtin_amdgcn_exp2f((sc[s][g] - mnew) * K2);
        psum += p;
        ptile[wave][row * 64 + s * 16 + (lane & 15)] = (__bf16)p;
      }
      tsum[g] = psum;
    }
    rowreduce_sum8(tsum);
#pragma unroll
    for (int g = 0; g < 8; ++g) lrun[g] = lrun[g] * scl[g] + tsum[g];
#pragma unroll
    for (int t = 0; t < 4; ++t)
#pragma unroll
      for (int g = 0; g < 8; ++g) acc[t][g] *= scl[g];

    v16bf pa0 = loadA16(&ptile[wave][0], 64, lane);       // keys j..j+31
    v16bf pa1 = loadA16(&ptile[wave][0] + 32, 64, lane);  // keys j+32..j+63
#pragma unroll
    for (int t = 0; t < 4; ++t) {
      const __bf16* Vb = VT + ((size_t)bh * HDIM + t * 16) * S_LEN + j;
      v16bf vb0 = loadA16(Vb, S_LEN, lane);
      v16bf vb1 = loadA16(Vb + 32, S_LEN, lane);
      acc[t] = wmma_bf16(pa1, vb1, wmma_bf16(pa0, vb0, acc[t]));
    }
  }

  const int b = bh >> 4, h = bh & 15;
#pragma unroll
  for (int t = 0; t < 4; ++t) {
    const int n = t * 16 + (lane & 15);
#pragma unroll
    for (int g = 0; g < 8; ++g) {
      const int mrow = g + ((lane >> 4) << 3);
      const float val = acc[t][g] / lrun[g];
      CTX[((size_t)b * S_LEN + q0 + mrow) * DMODEL + h * HDIM + n] = (__bf16)val;
    }
  }
}

extern "C" void kernel_launch(void* const* d_in, const int* in_sizes, int n_in,
                              void* d_out, int out_size, void* d_ws, size_t ws_size,
                              hipStream_t stream) {
  (void)in_sizes; (void)n_in; (void)out_size; (void)ws_size;
  const float* x  = (const float*)d_in[0];
  const float* Wq = (const float*)d_in[1];
  const float* bq = (const float*)d_in[2];
  const float* Wk = (const float*)d_in[3];
  const float* bk = (const float*)d_in[4];
  const float* Wv = (const float*)d_in[5];
  const float* bv = (const float*)d_in[6];
  const float* Wo = (const float*)d_in[7];
  const float* bo = (const float*)d_in[8];

  char* ws = (char*)d_ws;
  const size_t MB = 1ull << 20;
  __bf16* xh   = (__bf16*)(ws + 0 * MB);   // 8 MB  x in bf16
  __bf16* wqh  = (__bf16*)(ws + 8 * MB);   // 2 MB
  __bf16* wkh  = (__bf16*)(ws + 10 * MB);  // 2 MB
  __bf16* wvh  = (__bf16*)(ws + 12 * MB);  // 2 MB
  __bf16* woh  = (__bf16*)(ws + 14 * MB);  // 2 MB
  __bf16* qb   = (__bf16*)(ws + 16 * MB);  // 8 MB  [B,H,S,64]
  __bf16* kb   = (__bf16*)(ws + 24 * MB);  // 8 MB  [B,H,S,64]
  __bf16* vtb  = (__bf16*)(ws + 32 * MB);  // 8 MB  [B,H,64,S]
  __bf16* ctxh = (__bf16*)(ws + 40 * MB);  // 8 MB  [B,S,1024]

  cast_f32_bf16<<<2048, 256, 0, stream>>>(x, xh, MROWS * DMODEL);
  cast_f32_bf16<<<1024, 256, 0, stream>>>(Wq, wqh, DMODEL * DMODEL);
  cast_f32_bf16<<<1024, 256, 0, stream>>>(Wk, wkh, DMODEL * DMODEL);
  cast_f32_bf16<<<1024, 256, 0, stream>>>(Wv, wvh, DMODEL * DMODEL);
  cast_f32_bf16<<<1024, 256, 0, stream>>>(Wo, woh, DMODEL * DMODEL);

  gemm_bias<0><<<1024, 128, 0, stream>>>(xh, wqh, bq, qb);
  gemm_bias<0><<<1024, 128, 0, stream>>>(xh, wkh, bk, kb);
  gemm_bias<1><<<1024, 128, 0, stream>>>(xh, wvh, bv, vtb);

  flash_attn<<<dim3(32, 32), 128, 0, stream>>>(qb, kb, vtb, ctxh);

  gemm_bias<2><<<1024, 128, 0, stream>>>(ctxh, woh, bo, d_out);
}
